// Encoder_8753143349434
// MI455X (gfx1250) — matrix operand order repaired and checked by reading, not compiled, not run on hardware
//
#include <hip/hip_runtime.h>
#include <hip/hip_bf16.h>
#include <math.h>

// ---------------------------------------------------------------------------
// SwinV2 encoder for MI455X (gfx1250, wave32).  All GEMM-shaped math runs on
// v_wmma_f32_16x16x32_f16 (f16 operands, f32 accum).  Model is HBM-bandwidth
// bound (~160 GFLOP vs ~8-10 GB of activation traffic @ 23.3 TB/s), so all
// activation intermediates feeding matrix units are f16, and bias / GELU /
// LayerNorm / softmax / masks are fused into producer kernels.
// GEMM k-step is load-phase (A + NT B frags, one clause) then compute-phase
// (NT WMMAs) so WMMAs overlap outstanding loads via partial s_wait_loadcnt.
// ---------------------------------------------------------------------------

typedef __attribute__((ext_vector_type(16))) _Float16 v16h;
typedef __attribute__((ext_vector_type(8)))  _Float16 v8h;
typedef __attribute__((ext_vector_type(8)))  float    v8f;

#define DEV static __device__ __forceinline__

DEV v8f wmma_f16(v16h a, v16h b, v8f c) {
  return __builtin_amdgcn_wmma_f32_16x16x32_f16(false, a, false, b, (short)0, c,
                                                false, false);
}

// A fragment: 16x32 f16 tile, rows m0..m0+15, k block kb..kb+31, row-major lda.
// ISA layout: lanes 0-15 hold K=0..7 (VGPR0-3) and K=16..23 (VGPR4-7);
// lanes 16-31 hold K=8..15 and K=24..31.
DEV v16h load_a16(const _Float16* base, int lda, int m0, int kb, int lane) {
  int r = lane & 15, hf = lane >> 4;
  const _Float16* p = base + (size_t)(m0 + r) * lda + kb + hf * 8;
  v8h lo = *(const v8h*)p;
  v8h hi = *(const v8h*)(p + 16);
  return __builtin_shufflevector(lo, hi, 0, 1, 2, 3, 4, 5, 6, 7, 8, 9, 10, 11,
                                 12, 13, 14, 15);
}

// B fragment: 32x16 f16 tile, read from row-major [N,K] storage (weight layout
// W[out,in]).  Lanes 0-15 hold K=0..7,16..23, lanes 16-31 K=8..15,24..31; column = n0+lane%16.
DEV v16h load_b16(const _Float16* base, int ldb, int n0, int kb, int lane) {
  int r = lane & 15, hf = lane >> 4;
  const _Float16* p = base + (size_t)(n0 + r) * ldb + kb + hf * 8;
  v8h lo = *(const v8h*)p;
  v8h hi = *(const v8h*)(p + 16);
  return __builtin_shufflevector(lo, hi, 0, 1, 2, 3, 4, 5, 6, 7, 8, 9, 10, 11,
                                 12, 13, 14, 15);
}

// ---------------------------------------------------------------------------
// Generic WMMA GEMM:  out[M,N] = A[M,K](f16,row-major) * Bw[N,K](f16) + bias
// MODE 0: f32 out.  MODE 2: f16 out with exact GELU fused (fc1 path).
// Wave tile 16 x (16*NT); block = 8 waves => 128 x (16*NT) per workgroup.
// NT chosen by host so the grid tiles N exactly: no guards in the k-loop.
// ---------------------------------------------------------------------------
template <int MODE, int NT>
__global__ __launch_bounds__(256) void gemm_wmma(
    const _Float16* __restrict__ A, const _Float16* __restrict__ Bw,
    const float* __restrict__ bias, void* __restrict__ outp, int M, int N, int K) {
  int lane = threadIdx.x & 31;
  int wave = threadIdx.x >> 5;
  int m0 = blockIdx.y * 128 + wave * 16;
  int n0 = blockIdx.x * (16 * NT);
  if (m0 >= M) return;
  v8f zero = {0.f, 0.f, 0.f, 0.f, 0.f, 0.f, 0.f, 0.f};
  v8f acc[NT];
#pragma unroll
  for (int t = 0; t < NT; ++t) acc[t] = zero;
#pragma unroll 2
  for (int kb = 0; kb < K; kb += 32) {
    // load phase: issue all fragments for this k-step back-to-back so the
    // compute phase only needs partial loadcnt waits (WMMA/load overlap).
    v16h a = load_a16(A, K, m0, kb, lane);
    v16h b[NT];
#pragma unroll
    for (int t = 0; t < NT; ++t) b[t] = load_b16(Bw, K, n0 + 16 * t, kb, lane);
    // compute phase
#pragma unroll
    for (int t = 0; t < NT; ++t) acc[t] = wmma_f16(a, b[t], acc[t]);
  }
  int r = lane & 15, hf = lane >> 4;
#pragma unroll
  for (int t = 0; t < NT; ++t) {
    int n = n0 + 16 * t + r;
    float bs = bias ? bias[n] : 0.f;
#pragma unroll
    for (int i = 0; i < 8; ++i) {
      int m = m0 + i + 8 * hf;  // C/D layout: VGPR i -> row i (+8 for hi half)
      float v = acc[t][i] + bs;
      size_t idx = (size_t)m * N + n;
      if (MODE == 0) {
        ((float*)outp)[idx] = v;
      } else {
        float g = 0.5f * v * (1.f + erff(v * 0.70710678118f));  // exact GELU
        ((_Float16*)outp)[idx] = (_Float16)g;
      }
    }
  }
}

// ---------------------------------------------------------------------------
// Window partition (+cyclic shift) with f32->f16 conversion for the QKV GEMM.
// ---------------------------------------------------------------------------
__global__ __launch_bounds__(256) void win_partition(
    const float* __restrict__ x, _Float16* __restrict__ xw, int Bn, int H, int W,
    int C, int ws, int shift) {
  size_t tid = (size_t)blockIdx.x * 256 + threadIdx.x;
  size_t total = (size_t)Bn * H * W * C;
  if (tid >= total) return;
  int c = (int)(tid % C);
  size_t row = tid / C;
  int N = ws * ws;
  int t = (int)(row % N);
  size_t win = row / N;
  int wpr = W / ws, hpr = H / ws;
  int b = (int)(win / (hpr * wpr));
  int rem = (int)(win % (hpr * wpr));
  int wh = rem / wpr, ww = rem % wpr;
  int h = (wh * ws + t / ws + shift) % H;  // roll(x, -shift) then partition
  int w = (ww * ws + t % ws + shift) % W;
  xw[tid] = (_Float16)x[(((size_t)b * H + h) * W + w) * C + c];
}

// ---------------------------------------------------------------------------
// Per-(token,head): L2-normalize q,k; fold exp(min(logit_scale,ln100)) into q;
// emit f16 q/k ([M,C]) and transposed v ([win,head, d=32, tok(padded)]).
// One wave = one (token, head); lane = head-dim element.
// ---------------------------------------------------------------------------
__global__ __launch_bounds__(256) void qkv_prep(
    const float* __restrict__ qkv, const float* __restrict__ logit_scale,
    _Float16* __restrict__ qh, _Float16* __restrict__ kh, _Float16* __restrict__ vt,
    int M, int C, int nH, int N, int VST) {
  int g = blockIdx.x * 8 + (threadIdx.x >> 5);
  int lane = threadIdx.x & 31;
  int row = g / nH, head = g % nH;
  if (row >= M) return;
  const float* base = qkv + (size_t)row * 3 * C + head * 32;
  float q = base[lane];
  float k = base[C + lane];
  float v = base[2 * C + lane];
  float qs = q * q, ks = k * k;
#pragma unroll
  for (int off = 1; off < 32; off <<= 1) {
    qs += __shfl_xor(qs, off, 32);
    ks += __shfl_xor(ks, off, 32);
  }
  float qn = fmaxf(sqrtf(qs), 1e-12f), kn = fmaxf(sqrtf(ks), 1e-12f);
  float sc = __expf(fminf(logit_scale[head], 4.6051702f));
  qh[(size_t)row * C + head * 32 + lane] = (_Float16)(q / qn * sc);
  kh[(size_t)row * C + head * 32 + lane] = (_Float16)(k / kn);
  int tok = row % N;
  size_t win = row / N;
  vt[((win * nH + head) * 32 + lane) * (size_t)VST + tok] = (_Float16)v;
}

// ---------------------------------------------------------------------------
// Fused window attention.  WS=8: 4 waves per (win,head), each wave owns a
// 16-row strip of S.  WS=4: one wave per (win,head) (K padded 16->32 with
// zeros through LDS / pre-zeroed vt).  Softmax rows live across lanes 0..15 of
// each lane-half -> shfl_xor reductions.  S re-layouts C->A through LDS.
// ---------------------------------------------------------------------------
template <int WS>
__global__ __launch_bounds__(128) void attn_wmma(
    const _Float16* __restrict__ qh, const _Float16* __restrict__ kh,
    const _Float16* __restrict__ vt, _Float16* __restrict__ obuf,
    const float* __restrict__ rpb, int nWin, int nH, int Himg, int Wimg,
    int shift, int C) {
  constexpr int N = WS * WS;
  constexpr int NT = N / 16;
  constexpr int VST = (WS == 8) ? 64 : 32;   // padded K for S@V
  constexpr int SST = (WS == 8) ? 80 : 40;   // LDS row stride (f16)
  __shared__ alignas(32) _Float16 smem[4][16 * SST];
  int wave = threadIdx.x >> 5, lane = threadIdx.x & 31;
  int strip = blockIdx.x * 4 + wave;
  int whid = (WS == 8) ? (strip >> 2) : strip;
  int m0 = (WS == 8) ? ((strip & 3) << 4) : 0;
  int win = whid / nH, head = whid % nH;
  int r = lane & 15, hf = lane >> 4;
  v8f zero = {0.f, 0.f, 0.f, 0.f, 0.f, 0.f, 0.f, 0.f};

  // S = q_hat @ k_hat^T  (K = head_dim = 32, single WMMA k-step)
  const _Float16* qbp = qh + head * 32;
  const _Float16* kbp = kh + head * 32;
  v16h aq = load_a16(qbp, C, win * N + m0, 0, lane);
  v16h bk[NT];
#pragma unroll
  for (int t = 0; t < NT; ++t)
    bk[t] = load_b16(kbp, C, win * N + 16 * t, 0, lane);
  v8f s[NT];
#pragma unroll
  for (int t = 0; t < NT; ++t) s[t] = wmma_f16(aq, bk[t], zero);

  // relative-position bias + SW-MSA mask (region ids from window coords)
  int wpr = Wimg / WS;
  int rem = win % ((Himg / WS) * wpr);
  int wh = rem / wpr, ww = rem % wpr;
  const float* rp = rpb + (size_t)head * N * N;
  int creg[NT];
#pragma unroll
  for (int t = 0; t < NT; ++t) {
    int n = r + 16 * t;
    int hh2 = wh * WS + n / WS, wc2 = ww * WS + n % WS;
    int rh = (hh2 < Himg - WS) ? 0 : ((hh2 < Himg - shift) ? 1 : 2);
    int rw = (wc2 < Wimg - WS) ? 0 : ((wc2 < Wimg - shift) ? 1 : 2);
    creg[t] = rh * 3 + rw;
  }
#pragma unroll
  for (int i = 0; i < 8; ++i) {
    int m = m0 + i + 8 * hf;
    int hh2 = wh * WS + m / WS, wc2 = ww * WS + m % WS;
    int rh = (hh2 < Himg - WS) ? 0 : ((hh2 < Himg - shift) ? 1 : 2);
    int rw = (wc2 < Wimg - WS) ? 0 : ((wc2 < Wimg - shift) ? 1 : 2);
    int mreg = rh * 3 + rw;
#pragma unroll
    for (int t = 0; t < NT; ++t) {
      int n = r + 16 * t;
      float b2 = rp[(size_t)m * N + n];
      if (shift > 0 && mreg != creg[t]) b2 -= 100.f;
      s[t][i] += b2;
    }
  }

  // softmax over each row (row spans lanes 0..15 of this half across NT tiles)
#pragma unroll
  for (int i = 0; i < 8; ++i) {
    float mx = s[0][i];
#pragma unroll
    for (int t = 1; t < NT; ++t) mx = fmaxf(mx, s[t][i]);
#pragma unroll
    for (int off = 1; off < 16; off <<= 1) mx = fmaxf(mx, __shfl_xor(mx, off, 32));
    float sum = 0.f;
#pragma unroll
    for (int t = 0; t < NT; ++t) { float e = __expf(s[t][i] - mx); s[t][i] = e; sum += e; }
#pragma unroll
    for (int off = 1; off < 16; off <<= 1) sum += __shfl_xor(sum, off, 32);
    float inv = 1.f / sum;
#pragma unroll
    for (int t = 0; t < NT; ++t) s[t][i] *= inv;
  }

  // C-layout -> A-layout via per-wave LDS patch (zero-pad K for WS==4)
  _Float16* sp = &smem[wave][0];
#pragma unroll
  for (int i = 0; i < 8; ++i) {
    int lr = i + 8 * hf;
#pragma unroll
    for (int t = 0; t < NT; ++t) sp[lr * SST + r + 16 * t] = (_Float16)s[t][i];
    if (WS == 4) sp[lr * SST + 16 + r] = (_Float16)0.f;
  }
  __syncthreads();

  // O = S @ V  (V pre-transposed: [d=32, tok(VST)])
  const _Float16* vbp = vt + ((size_t)win * nH + head) * 32 * VST;
  v8f o0 = zero, o1 = zero;
#pragma unroll
  for (int kb2 = 0; kb2 < VST; kb2 += 32) {
    v16h as = load_a16(sp, SST, 0, kb2, lane);
    v16h b0 = load_b16(vbp, VST, 0, kb2, lane);
    v16h b1 = load_b16(vbp, VST, 16, kb2, lane);
    o0 = wmma_f16(as, b0, o0);
    o1 = wmma_f16(as, b1, o1);
  }
#pragma unroll
  for (int i = 0; i < 8; ++i) {
    size_t orow = (size_t)win * N + m0 + i + 8 * hf;
    obuf[orow * C + head * 32 + r] = (_Float16)o0[i];
    obuf[orow * C + head * 32 + 16 + r] = (_Float16)o1[i];
  }
}

// ---------------------------------------------------------------------------
// x = x + LayerNorm(y)*g + b, with optional window-reverse(+roll) gather of y
// and optional f16 mirror of the result.  One wave per token.
// ---------------------------------------------------------------------------
__global__ __launch_bounds__(256) void residual_ln(
    const float* __restrict__ y, float* __restrict__ x,
    const float* __restrict__ gamma, const float* __restrict__ beta,
    _Float16* __restrict__ xh, int Bn, int H, int W, int C, int ws, int shift,
    int winmap) {
  int tok = blockIdx.x * 8 + (threadIdx.x >> 5);
  int lane = threadIdx.x & 31;
  int M = Bn * H * W;
  if (tok >= M) return;
  size_t yrow = tok;
  if (winmap) {
    int b = tok / (H * W);
    int rem = tok % (H * W);
    int h = rem / W, w = rem % W;
    int hh = (h - shift + H) % H, wc = (w - shift + W) % W;
    int wpr = W / ws, N = ws * ws;
    yrow = (((size_t)b * (H / ws) + hh / ws) * wpr + wc / ws) * N +
           (hh % ws) * ws + (wc % ws);
  }
  int nc = C >> 5;
  float yv[8];
  float s = 0.f;
  for (int k2 = 0; k2 < nc; ++k2) {
    yv[k2] = y[yrow * C + lane + 32 * k2];
    s += yv[k2];
  }
  for (int off = 1; off < 32; off <<= 1) s += __shfl_xor(s, off, 32);
  float mean = s / C;
  float vs = 0.f;
  for (int k2 = 0; k2 < nc; ++k2) { float d = yv[k2] - mean; vs += d * d; }
  for (int off = 1; off < 32; off <<= 1) vs += __shfl_xor(vs, off, 32);
  float rstd = rsqrtf(vs / C + 1e-5f);
  for (int k2 = 0; k2 < nc; ++k2) {
    int c = lane + 32 * k2;
    float o = x[(size_t)tok * C + c] + (yv[k2] - mean) * rstd * gamma[c] + beta[c];
    x[(size_t)tok * C + c] = o;
    if (xh) xh[(size_t)tok * C + c] = (_Float16)o;
  }
}

// --------------------------- small utility kernels --------------------------
__global__ void cvt_f16k(const float* __restrict__ in, _Float16* __restrict__ o, int n) {
  int i = blockIdx.x * 256 + threadIdx.x;
  if (i < n) o[i] = (_Float16)in[i];
}

__global__ void qkv_bias_assemble(const float* __restrict__ qb,
                                  const float* __restrict__ vb,
                                  float* __restrict__ o, int C) {
  int i = blockIdx.x * 256 + threadIdx.x;
  if (i < C) o[i] = qb[i];
  else if (i < 2 * C) o[i] = 0.f;
  else if (i < 3 * C) o[i] = vb[i - 2 * C];
}

// SwinV2 continuous position bias MLP: log-spaced coords -> 512 relu -> nH
__global__ void cpb_mlp(const float* __restrict__ w1, const float* __restrict__ b1,
                        const float* __restrict__ w2, float* __restrict__ bt,
                        int ws, int nH) {
  int L = 2 * ws - 1, R = L * L;
  int tid = blockIdx.x * 64 + threadIdx.x;
  if (tid >= R * nH) return;
  int row = tid / nH, h = tid % nH;
  float m = (float)(ws - 1); if (m < 1.f) m = 1.f;
  float v0 = (float)(row / L - (ws - 1)) / m * 8.f;
  float v1 = (float)(row % L - (ws - 1)) / m * 8.f;
  float t0 = copysignf(log2f(fabsf(v0) + 1.f) / 3.f, v0);
  float t1 = copysignf(log2f(fabsf(v1) + 1.f) / 3.f, v1);
  float acc = 0.f;
  for (int j = 0; j < 512; ++j) {
    float hv = fmaxf(w1[2 * j] * t0 + w1[2 * j + 1] * t1 + b1[j], 0.f);
    acc += w2[h * 512 + j] * hv;
  }
  bt[row * nH + h] = acc;
}

__global__ void rpb_build(const float* __restrict__ bt, float* __restrict__ rpb,
                          int ws, int nH) {
  int N = ws * ws;
  int tid = blockIdx.x * 256 + threadIdx.x;
  if (tid >= nH * N * N) return;
  int h = tid / (N * N);
  int rem = tid % (N * N);
  int i = rem / N, j = rem % N;
  int L = 2 * ws - 1;
  int idx = (i / ws - j / ws + ws - 1) * L + (i % ws - j % ws + ws - 1);
  rpb[tid] = 16.f / (1.f + __expf(-bt[idx * nH + h]));
}

// conv weight HWIO [4,4,C,2C] -> f16 [N=2C, K=16C] (weight layout for GEMM B)
__global__ void convw_t(const float* __restrict__ w, _Float16* __restrict__ wt,
                        int K, int Nc) {
  size_t tid = (size_t)blockIdx.x * 256 + threadIdx.x;
  if (tid >= (size_t)K * Nc) return;
  int n = (int)(tid % Nc);
  size_t k = tid / Nc;
  wt[(size_t)n * K + k] = (_Float16)w[k * Nc + n];
}

// implicit im2col for 4x4 stride-2 pad-1 conv; one thread per (out-pixel, tap)
__global__ void im2col(const float* __restrict__ x, _Float16* __restrict__ col,
                       int Bn, int H, int W, int C, int Ho, int Wo) {
  size_t tid = (size_t)blockIdx.x * 256 + threadIdx.x;
  size_t tot = (size_t)Bn * Ho * Wo * 16;
  if (tid >= tot) return;
  int kk = (int)(tid % 16);
  size_t op = tid / 16;
  int ow = (int)(op % Wo);
  size_t t2 = op / Wo;
  int oh = (int)(t2 % Ho);
  int b = (int)(t2 / Ho);
  int ih = oh * 2 - 1 + kk / 4, iw = ow * 2 - 1 + kk % 4;
  _Float16* dst = col + (op * 16 + kk) * (size_t)C;
  if (ih >= 0 && ih < H && iw >= 0 && iw < W) {
    const float* src = x + (((size_t)b * H + ih) * W + iw) * C;
    for (int c = 0; c < C; ++c) dst[c] = (_Float16)src[c];
  } else {
    for (int c = 0; c < C; ++c) dst[c] = (_Float16)0.f;
  }
}

// ---------------------------------------------------------------------------
extern "C" void kernel_launch(void* const* d_in, const int* in_sizes, int n_in,
                              void* d_out, int out_size, void* d_ws, size_t ws_size,
                              hipStream_t stream) {
  (void)out_size; (void)ws_size;
  const int Bn = 8, HIN = 256, WIN_ = 288, CB = 32;
  const int WSZ[4] = {8, 8, 4, 4};
  const size_t MC = (size_t)Bn * HIN * WIN_ * CB;  // 18,874,368 (stage-0 M*C)

  // ---- workspace carve (~620 MB, stage-0 sizes dominate everywhere) ----
  char* wsp = (char*)d_ws;
  size_t off = 0;
  auto carve = [&](size_t bytes) -> char* {
    char* p = wsp + off;
    off += (bytes + 255) & ~(size_t)255;
    return p;
  };
  _Float16* XW = (_Float16*)carve(MC * 2);       // window-partitioned x (f16)
  _Float16* QH = (_Float16*)carve(MC * 2);       // normalized+scaled q (f16)
  _Float16* KH = (_Float16*)carve(MC * 2);       // normalized k (f16)
  _Float16* VT = (_Float16*)carve(MC * 2);       // transposed v (f16, padded K)
  _Float16* OB = (_Float16*)carve(MC * 2);       // attention output (f16)
  float*    YB = (float*)carve(MC * 4);          // proj output (f32)
  _Float16* XH = (_Float16*)carve(MC * 2);       // f16 mirror of x (fc1 input)
  char*     GBr = carve(MC * 8);                 // gbuf f16 [M,4C] / im2col / qkv lo
  float*    HB = (float*)carve(MC * 4);          // fc2 out f32 / qkv hi (alias)
  float*    XB = (float*)carve(MC * 4);          // current activation x (f32)
  _Float16* WQ = (_Float16*)carve((size_t)3 * 256 * 256 * 2);
  _Float16* WP = (_Float16*)carve((size_t)256 * 256 * 2);
  _Float16* W1 = (_Float16*)carve((size_t)1024 * 256 * 2);
  _Float16* W2 = (_Float16*)carve((size_t)1024 * 256 * 2);
  _Float16* WC = (_Float16*)carve((size_t)4096 * 512 * 2);
  float*    QB = (float*)carve(768 * 4);
  float*    BT = (float*)carve(225 * 8 * 4);
  float*    RPB = (float*)carve(8192 * 4);
  float*    QKV32 = (float*)GBr;     // aliases GB(8*MC)+HB(4*MC) = 12*MC bytes
  _Float16* GB = (_Float16*)GBr;

  // ---- input index mapping: dict insertion order (x first) with fallback to
  //      jax sorted-key order (params first, x last), detected via in_sizes ----
  int xIdx = 0;
  for (int i = 0; i < n_in; ++i)
    if (in_sizes[i] == (int)MC) { xIdx = i; break; }
  bool alpha = (xIdx != 0);
  const float* xin = (const float*)d_in[xIdx];
  // field order (ours): qkv_w q_bias v_bias logit_scale cpb_w1 cpb_b1 cpb_w2
  //                     proj_w proj_b n1g n1b fc1_w fc1_b fc2_w fc2_b n2g n2b
  static const int insMap[17] = {0,1,2,3,4,5,6,7,8,9,10,11,12,13,14,15,16};
  static const int alpMap[17] = {15,14,16,7,1,0,2,13,12,9,8,4,3,6,5,11,10};
  const int* fmap = alpha ? alpMap : insMap;

  // ---- output regions: (final, skip0..3) concatenated ----
  float* outF = (float*)d_out;
  float* skipPtr[4];
  size_t acc = (size_t)Bn * 16 * 18 * 512;  // final tensor size
  for (int i = 0; i < 4; ++i) {
    skipPtr[i] = outF + acc;
    acc += (size_t)Bn * (HIN >> i) * (WIN_ >> i) * (CB << i);
  }

  // GEMM dispatch: NT=4 (16x64 wave tile) when N%64==0, else NT=2 (16x32)
  auto gemm = [&](int mode, const _Float16* A, const _Float16* Bw,
                  const float* bias, void* out, int M, int N, int K) {
    if (N % 64 == 0) {
      dim3 g(N / 64, M / 128);
      if (mode == 0) gemm_wmma<0, 4><<<g, 256, 0, stream>>>(A, Bw, bias, out, M, N, K);
      else           gemm_wmma<2, 4><<<g, 256, 0, stream>>>(A, Bw, bias, out, M, N, K);
    } else {
      dim3 g(N / 32, M / 128);
      if (mode == 0) gemm_wmma<0, 2><<<g, 256, 0, stream>>>(A, Bw, bias, out, M, N, K);
      else           gemm_wmma<2, 2><<<g, 256, 0, stream>>>(A, Bw, bias, out, M, N, K);
    }
  };

  hipMemcpyAsync(XB, xin, MC * 4, hipMemcpyDeviceToDevice, stream);

  for (int s = 0; s < 4; ++s) {
    int H = HIN >> s, W = WIN_ >> s, C = CB << s;
    int nH = C / 32, ws2 = WSZ[s], N = ws2 * ws2;
    int M = Bn * H * W;
    int nWin = M / N;
    int base = (alpha ? 0 : 1) + s * 36;

    for (int blk = 0; blk < 2; ++blk) {
      int pb = base + blk * 17;
      const float* P[17];
      for (int j = 0; j < 17; ++j) P[j] = (const float*)d_in[pb + fmap[j]];
      const float *qkv_w = P[0], *q_bias = P[1], *v_bias = P[2], *lsc = P[3];
      const float *cw1 = P[4], *cb1 = P[5], *cw2 = P[6];
      const float *pw = P[7], *pbias = P[8], *n1g = P[9], *n1b = P[10];
      const float *f1w = P[11], *f1b = P[12], *f2w = P[13], *f2b = P[14];
      const float *n2g = P[15], *n2b = P[16];
      int shift = blk ? ws2 / 2 : 0;

      // weight prep (f16 conversions; all < 6 MB -> L2 resident)
      cvt_f16k<<<(3 * C * C + 255) / 256, 256, 0, stream>>>(qkv_w, WQ, 3 * C * C);
      cvt_f16k<<<(C * C + 255) / 256, 256, 0, stream>>>(pw, WP, C * C);
      cvt_f16k<<<(4 * C * C + 255) / 256, 256, 0, stream>>>(f1w, W1, 4 * C * C);
      cvt_f16k<<<(4 * C * C + 255) / 256, 256, 0, stream>>>(f2w, W2, 4 * C * C);
      qkv_bias_assemble<<<(3 * C + 255) / 256, 256, 0, stream>>>(q_bias, v_bias, QB, C);
      int L = 2 * ws2 - 1, R = L * L;
      cpb_mlp<<<(R * nH + 63) / 64, 64, 0, stream>>>(cw1, cb1, cw2, BT, ws2, nH);
      rpb_build<<<(nH * N * N + 255) / 256, 256, 0, stream>>>(BT, RPB, ws2, nH);

      // attention path
      size_t tot = (size_t)M * C;
      win_partition<<<(unsigned)((tot + 255) / 256), 256, 0, stream>>>(
          XB, XW, Bn, H, W, C, ws2, shift);
      gemm(0, XW, WQ, QB, QKV32, M, 3 * C, C);
      int VST = (ws2 == 8) ? 64 : 32;
      if (ws2 == 4)
        hipMemsetAsync(VT, 0, (size_t)nWin * nH * 32 * VST * 2, stream);
      qkv_prep<<<(M * nH) / 8, 256, 0, stream>>>(QKV32, lsc, QH, KH, VT, M, C, nH, N, VST);
      if (ws2 == 8)
        attn_wmma<8><<<nWin * nH, 128, 0, stream>>>(QH, KH, VT, OB, RPB, nWin, nH,
                                                    H, W, shift, C);
      else
        attn_wmma<4><<<(nWin * nH) / 4, 128, 0, stream>>>(QH, KH, VT, OB, RPB,
                                                          nWin, nH, H, W, shift, C);
      gemm(0, OB, WP, pbias, YB, M, C, C);
      // window-reverse + roll + residual + post-LN (also emits f16 x for fc1)
      residual_ln<<<M / 8, 256, 0, stream>>>(YB, XB, n1g, n1b, XH, Bn, H, W, C,
                                             ws2, shift, 1);
      // MLP: fc1+GELU fused (f16 out), fc2, residual + post-LN
      gemm(2, XH, W1, f1b, GB, M, 4 * C, C);
      gemm(0, GB, W2, f2b, HB, M, C, 4 * C);
      residual_ln<<<M / 8, 256, 0, stream>>>(HB, XB, n2g, n2b, (_Float16*)nullptr,
                                             Bn, H, W, C, 0, 0, 0);
    }

    // stage skip output
    hipMemcpyAsync(skipPtr[s], XB, (size_t)M * C * 4, hipMemcpyDeviceToDevice, stream);

    // 4x4 / stride-2 / pad-1 conv as im2col + WMMA GEMM
    const float* convw = (const float*)d_in[base + 34 + (alpha ? 1 : 0)];
    const float* convb = (const float*)d_in[base + (alpha ? 34 : 35)];
    int Ho = H / 2, Wo = W / 2, Co = 2 * C, Kc = 16 * C, Mo = Bn * Ho * Wo;
    convw_t<<<(unsigned)(((size_t)Kc * Co + 255) / 256), 256, 0, stream>>>(convw, WC, Kc, Co);
    im2col<<<(unsigned)(((size_t)Mo * 16 + 255) / 256), 256, 0, stream>>>(
        XB, GB, Bn, H, W, C, Ho, Wo);
    float* co = (s == 3) ? outF : XB;
    gemm(0, GB, WC, convb, co, Mo, Co, Kc);
  }
}
